// Attention_21105469292726
// MI455X (gfx1250) — matrix operand order repaired and checked by reading, not compiled, not run on hardware
//
#include <hip/hip_runtime.h>
#include <stdint.h>

// ---------------------------------------------------------------------------
// Problem constants (match reference)
// ---------------------------------------------------------------------------
#define TSEQ   4096
#define DMODEL 2048
#define NH     16
#define DH     128
#define D3     (3 * DMODEL)

typedef __attribute__((ext_vector_type(16))) __bf16 v16bf;
typedef __attribute__((ext_vector_type(8)))  float  v8f;
typedef __attribute__((ext_vector_type(4)))  unsigned int v4u;
typedef __attribute__((ext_vector_type(8)))  int    v8i;
typedef __attribute__((ext_vector_type(4)))  int    v4i;

union Frag {            // one WMMA A/B operand = 8 VGPRs = 32 bytes / lane
    v16bf v;
    uint4 q[2];
};

__device__ __forceinline__ v8f wmma_bf16(v16bf a, v16bf b, v8f c) {
    return __builtin_amdgcn_wmma_f32_16x16x32_bf16(
        /*neg_a=*/false, a, /*neg_b=*/false, b,
        /*c_mod=*/(short)0, c, /*reuse_a=*/false, /*reuse_b=*/false);
}

// ---------------------------------------------------------------------------
// Tensor Data Mover descriptors (CDNA5 ISA)
// 2-D tile of a row-major tensor of 2-byte elements -> contiguous LDS rows.
// ---------------------------------------------------------------------------
__device__ __forceinline__ v4u tdm_g0(unsigned lds_off, const void* gaddr) {
    unsigned long long ga = (unsigned long long)(uintptr_t)gaddr;
    v4u g;
    g[0] = 1u;                                   // count=1, user descriptor
    g[1] = lds_off;                              // lds_addr [63:32]
    g[2] = (unsigned)(ga & 0xffffffffull);       // global_addr lo
    g[3] = (unsigned)((ga >> 32) & 0x01ffffffull) | (2u << 30);  // hi | type=2
    return g;
}

__device__ __forceinline__ v8i tdm_g1(unsigned dim0, unsigned dim1,
                                      unsigned tile0, unsigned tile1,
                                      unsigned long long stride0) {
    v8i g;
    g[0] = 1 << 16;                              // data_size = 1 -> 2 bytes
    g[1] = (int)((dim0 & 0xffffu) << 16);        // tensor_dim0[15:0] @ 63:48
    g[2] = (int)(((dim0 >> 16) & 0xffffu) | ((dim1 & 0xffffu) << 16));
    g[3] = (int)(((dim1 >> 16) & 0xffffu) | ((tile0 & 0xffffu) << 16));
    g[4] = (int)(tile1 & 0xffffu);               // tile_dim1 @ 143:128
    g[5] = (int)(stride0 & 0xffffffffull);       // tensor_dim0_stride lo
    g[6] = (int)((stride0 >> 32) & 0xffffull);   // stride hi
    g[7] = 0;
    return g;
}

__device__ __forceinline__ void tdm_load(v4u g0, v8i g1) {
    v4i z4 = {0, 0, 0, 0};
#if defined(__clang_major__) && (__clang_major__ >= 23)
    v8i z8 = {0, 0, 0, 0, 0, 0, 0, 0};
    __builtin_amdgcn_tensor_load_to_lds(g0, g1, z4, z4, z8, 0);
#else
    __builtin_amdgcn_tensor_load_to_lds(g0, g1, z4, z4, 0);
#endif
}

__device__ __forceinline__ unsigned lds_off(const void* p) {
    return (unsigned)(uintptr_t)p;   // generic LDS pointer: low 32 bits = offset
}

// ---------------------------------------------------------------------------
// Batched bf16 transpose:  dst[c][r] = src[r][c]
// ---------------------------------------------------------------------------
__global__ __launch_bounds__(256)
void transpose_bf16(const __bf16* __restrict__ src, __bf16* __restrict__ dst,
                    int R, int C, int sstride, int dstride,
                    long long sbatch, long long dbatch) {
    __shared__ __bf16 tile[32][33];
    const __bf16* s = src + (long long)blockIdx.z * sbatch;
    __bf16*       d = dst + (long long)blockIdx.z * dbatch;
    const int c0 = blockIdx.x * 32, r0 = blockIdx.y * 32;
    const int tx = threadIdx.x, ty = threadIdx.y;     // (32, 8)
#pragma unroll
    for (int j = 0; j < 4; ++j)
        tile[ty + j * 8][tx] = s[(size_t)(r0 + ty + j * 8) * sstride + c0 + tx];
    __syncthreads();
#pragma unroll
    for (int j = 0; j < 4; ++j)
        d[(size_t)(c0 + ty + j * 8) * dstride + r0 + tx] = tile[tx][ty + j * 8];
}

// ---------------------------------------------------------------------------
// bf16 WMMA GEMM: C[M,N] = A[M,K] * Bt[N,K]^T, f32 accumulate, bf16 out.
// Block tile 128x256, 8 waves, wave tile 64x64 (16 WMMA accumulators).
// Tiles staged by TDM (tensor_load_to_lds), double-buffered via TENSORcnt.
// ---------------------------------------------------------------------------
__global__ __launch_bounds__(256)
void gemm_bf16_wmma(const __bf16* __restrict__ A, const __bf16* __restrict__ Bt,
                    __bf16* __restrict__ C, int M, int N, int K) {
    __shared__ __bf16 As[2][128 * 32];   // [m][k]  16 KB
    __shared__ __bf16 Bs[2][256 * 32];   // [n][k]  32 KB
    const int tid  = threadIdx.x;
    const int lane = tid & 31;
    const int wave = tid >> 5;
    const int ln   = lane & 15;
    const bool hi  = lane >= 16;
    const int wm = wave >> 2;            // 0..1 -> 64-row strip
    const int wn = wave & 3;             // 0..3 -> 64-col strip
    const int m0 = blockIdx.y * 128;
    const int n0 = blockIdx.x * 256;
    const int kba = hi ? 8 : 0;          // A-fragment K base (ISA layout)
    const int kbb = hi ? 16 : 0;         // unused; B fragments also load with kba

    v8f acc[4][4];
#pragma unroll
    for (int ms = 0; ms < 4; ++ms)
#pragma unroll
        for (int ns = 0; ns < 4; ++ns)
            acc[ms][ns] = (v8f){0.f, 0.f, 0.f, 0.f, 0.f, 0.f, 0.f, 0.f};

    const v8i g1A = tdm_g1(K, M, 32, 128, (unsigned long long)K);
    const v8i g1B = tdm_g1(K, N, 32, 256, (unsigned long long)K);
    const int nsteps = K >> 5;

    if (wave == 0) {   // prologue: stage k-step 0 into buffer 0
        tdm_load(tdm_g0(lds_off(&As[0][0]), &A[(size_t)m0 * K]), g1A);
        tdm_load(tdm_g0(lds_off(&Bs[0][0]), &Bt[(size_t)n0 * K]), g1B);
    }

    for (int s = 0; s < nsteps; ++s) {
        const int buf = s & 1;
        if (wave == 0) {
            if (s + 1 < nsteps) {
                const int k1 = (s + 1) * 32;
                tdm_load(tdm_g0(lds_off(&As[buf ^ 1][0]),
                                &A[(size_t)m0 * K + k1]), g1A);
                tdm_load(tdm_g0(lds_off(&Bs[buf ^ 1][0]),
                                &Bt[(size_t)n0 * K + k1]), g1B);
                __builtin_amdgcn_s_wait_tensorcnt((short)2);  // current pair done
            } else {
                __builtin_amdgcn_s_wait_tensorcnt((short)0);
            }
        }
        __syncthreads();

        const __bf16* Ab = As[buf];
        const __bf16* Bb = Bs[buf];
        Frag bfr[4];
#pragma unroll
        for (int ns = 0; ns < 4; ++ns) {
            int row = wn * 64 + ns * 16 + ln;
            bfr[ns].q[0] = *(const uint4*)&Bb[row * 32 + kba];
            bfr[ns].q[1] = *(const uint4*)&Bb[row * 32 + 16 + kba];
        }
#pragma unroll
        for (int ms = 0; ms < 4; ++ms) {
            Frag af;
            int row = wm * 64 + ms * 16 + ln;
            af.q[0] = *(const uint4*)&Ab[row * 32 + kba];
            af.q[1] = *(const uint4*)&Ab[row * 32 + 16 + kba];
#pragma unroll
            for (int ns = 0; ns < 4; ++ns)
                acc[ms][ns] = wmma_bf16(af.v, bfr[ns].v, acc[ms][ns]);
        }
        __syncthreads();   // all waves done reading buf before it is re-staged
    }

    // C/D layout: lane<16 -> rows 0..7, lane>=16 -> rows 8..15 of each tile
#pragma unroll
    for (int ms = 0; ms < 4; ++ms)
#pragma unroll
        for (int ns = 0; ns < 4; ++ns) {
            int gn = n0 + wn * 64 + ns * 16 + ln;
#pragma unroll
            for (int r = 0; r < 8; ++r) {
                int gm = m0 + wm * 64 + ms * 16 + (hi ? 8 + r : r);
                C[(size_t)gm * N + gn] = (__bf16)acc[ms][ns][r];
            }
        }
}

// ---------------------------------------------------------------------------
// RoPE: q,k = qkv[:, :2048] / qkv[:, 2048:4096] rotated, emitted as bf16
// ---------------------------------------------------------------------------
__global__ __launch_bounds__(256)
void rope_kernel(const __bf16* __restrict__ qkv,
                 const float* __restrict__ cosb, const float* __restrict__ sinb,
                 __bf16* __restrict__ qr, __bf16* __restrict__ kr) {
    int idx = blockIdx.x * blockDim.x + threadIdx.x;   // over T*D
    int t = idx >> 11;
    int c = idx & 2047;
    int d = c & 127;
    float cs = cosb[t * DH + d];
    float sn = sinb[t * DH + d];
    const __bf16* row = qkv + (size_t)t * D3;
    int cpart = (d < 64) ? c + 64 : c - 64;
    float sgn = (d < 64) ? -1.f : 1.f;

    float qv = (float)row[c];
    float qrot = sgn * (float)row[cpart];
    qr[(size_t)t * DMODEL + c] = (__bf16)(qv * cs + qrot * sn);

    float kv = (float)row[DMODEL + c];
    float krot = sgn * (float)row[DMODEL + cpart];
    kr[(size_t)t * DMODEL + c] = (__bf16)(kv * cs + krot * sn);
}

// ---------------------------------------------------------------------------
// Flash attention (per head). Block = 4 waves = 64 queries; wave owns 16.
// S^T = K * Q^T (softmax stats per-lane), P via LDS to B-layout,
// O^T = V^T * P^T. K / V^T tiles staged by double-buffered TDM.
// ---------------------------------------------------------------------------
__global__ __launch_bounds__(128)
void flash_attn(const __bf16* __restrict__ qr, const __bf16* __restrict__ kr,
                const __bf16* __restrict__ vt, const unsigned char* __restrict__ mask,
                __bf16* __restrict__ obuf) {
    const int h    = blockIdx.y;
    const int qblk = blockIdx.x;          // 64-query block
    const int tid  = threadIdx.x;
    const int wave = tid >> 5, lane = tid & 31;
    const int ln   = lane & 15;
    const bool hi  = lane >= 16;
    const int kba  = hi ? 8 : 0;
    const int kbb  = hi ? 16 : 0;

    __shared__ __bf16 Ks[2][32 * DH];     // [key][dh]   16 KB
    __shared__ __bf16 Vts[2][DH * 32];    // [dh][key]   16 KB
    __shared__ __bf16 Ps[4][16 * 32];     // per-wave P^T [query][key] 4 KB

    const int q0 = qblk * 64 + wave * 16;
    const int t  = q0 + ln;
    const unsigned int mt = mask[t];

    Frag qf[4];
    const __bf16* qrow = qr + (size_t)t * DMODEL + h * DH;
#pragma unroll
    for (int c = 0; c < 4; ++c) {
        qf[c].q[0] = *(const uint4*)&qrow[c * 32 + kba];
        qf[c].q[1] = *(const uint4*)&qrow[c * 32 + 16 + kba];
    }

    v8f acc[8];
#pragma unroll
    for (int d = 0; d < 8; ++d)
        acc[d] = (v8f){0.f, 0.f, 0.f, 0.f, 0.f, 0.f, 0.f, 0.f};
    float m_run = -1e30f, l_run = 0.f;
    const float scale = 0.08838834764831845f;   // 1/sqrt(128)

    // TDM descriptors: K tensor (T x D), tile 32 rows x 128 cols;
    //                  V^T tensor (DH x T), tile 128 rows x 32 cols.
    const v8i g1K = tdm_g1(DMODEL, TSEQ, DH, 32, DMODEL);
    const v8i g1V = tdm_g1(TSEQ, DH, 32, DH, TSEQ);
    const __bf16* kbase = kr + h * DH;
    const __bf16* vbase = vt + (size_t)h * DH * TSEQ;

    const int nblocks = (qblk + 1) * 2;         // causal 32-key blocks
    if (wave == 0) {   // prologue: stage block 0 into buffer 0
        tdm_load(tdm_g0(lds_off(&Ks[0][0]), kbase), g1K);
        tdm_load(tdm_g0(lds_off(&Vts[0][0]), vbase), g1V);
    }

    for (int sb = 0; sb < nblocks; ++sb) {
        const int s0 = sb * 32;
        const int buf = sb & 1;
        if (wave == 0) {
            if (sb + 1 < nblocks) {
                const int s1 = s0 + 32;
                tdm_load(tdm_g0(lds_off(&Ks[buf ^ 1][0]),
                                kbase + (size_t)s1 * DMODEL), g1K);
                tdm_load(tdm_g0(lds_off(&Vts[buf ^ 1][0]), vbase + s1), g1V);
                __builtin_amdgcn_s_wait_tensorcnt((short)2);
            } else {
                __builtin_amdgcn_s_wait_tensorcnt((short)0);
            }
        }
        __syncthreads();
        const __bf16* Kb = Ks[buf];
        const __bf16* Vb = Vts[buf];

        // ---- S^T = K * Q^T : two 16(key) x 16(query) tiles ----
        v8f st[2];
        st[0] = (v8f){0.f, 0.f, 0.f, 0.f, 0.f, 0.f, 0.f, 0.f};
        st[1] = (v8f){0.f, 0.f, 0.f, 0.f, 0.f, 0.f, 0.f, 0.f};
#pragma unroll
        for (int mh = 0; mh < 2; ++mh) {
            int row = mh * 16 + ln;
#pragma unroll
            for (int c = 0; c < 4; ++c) {
                Frag a;
                a.q[0] = *(const uint4*)&Kb[row * DH + c * 32 + kba];
                a.q[1] = *(const uint4*)&Kb[row * DH + c * 32 + 16 + kba];
                st[mh] = wmma_bf16(a.v, qf[c].v, st[mh]);
            }
        }

        // ---- mask + online softmax (per-lane = per query column) ----
        unsigned long long ms0 = *(const unsigned long long*)(mask + s0 + kba);
        unsigned long long ms1 = *(const unsigned long long*)(mask + s0 + 16 + kba);
        float p[16];
        bool  ok[16];
        float rmax = -1e30f;
#pragma unroll
        for (int r = 0; r < 8; ++r) {
            int sa0 = s0 + kba + r;
            int sa1 = s0 + 16 + kba + r;
            unsigned int mk0 = (unsigned int)((ms0 >> (8 * r)) & 0xffu);
            unsigned int mk1 = (unsigned int)((ms1 >> (8 * r)) & 0xffu);
            ok[r]     = ((((mk0 & mt) != 0u) || (sa0 == t)) && (sa0 <= t));
            ok[8 + r] = ((((mk1 & mt) != 0u) || (sa1 == t)) && (sa1 <= t));
            p[r]     = ok[r]     ? st[0][r] * scale : -1e30f;
            p[8 + r] = ok[8 + r] ? st[1][r] * scale : -1e30f;
            rmax = fmaxf(rmax, fmaxf(p[r], p[8 + r]));
        }
        rmax = fmaxf(rmax, __shfl_xor(rmax, 16, 32));
        float m_new = fmaxf(m_run, rmax);
        float fac   = __expf(m_run - m_new);
        float psum  = 0.f;
#pragma unroll
        for (int i = 0; i < 16; ++i) {
            p[i] = ok[i] ? __expf(p[i] - m_new) : 0.f;
            psum += p[i];
        }
        l_run = l_run * fac + psum;
        m_run = m_new;
#pragma unroll
        for (int d = 0; d < 8; ++d)
            acc[d] *= fac;

        // ---- bounce P^T through LDS into B-fragment layout ----
        __bf16* prow = &Ps[wave][ln * 32];
#pragma unroll
        for (int r = 0; r < 8; ++r) {
            prow[kba + r]      = (__bf16)p[r];
            prow[16 + kba + r] = (__bf16)p[8 + r];
        }
        Frag pf;
        pf.q[0] = *(const uint4*)&Ps[wave][ln * 32 + kba];
        pf.q[1] = *(const uint4*)&Ps[wave][ln * 32 + 16 + kba];

        // ---- O^T += V^T * P^T : 8 dh-tiles, K = 32 keys in one WMMA ----
#pragma unroll
        for (int d = 0; d < 8; ++d) {
            Frag a;
            int row = d * 16 + ln;
            a.q[0] = *(const uint4*)&Vb[row * 32 + kba];
            a.q[1] = *(const uint4*)&Vb[row * 32 + 16 + kba];
            acc[d] = wmma_bf16(a.v, pf.v, acc[d]);
        }
        __syncthreads();   // all waves done with buf before re-staging
    }

    // ---- epilogue: combine halves of l, normalize, store O (bf16) ----
    float l_tot = l_run + __shfl_xor(l_run, 16, 32);
    float rinv  = 1.0f / l_tot;
#pragma unroll
    for (int d = 0; d < 8; ++d) {
        __bf16 ob[8];
#pragma unroll
        for (int r = 0; r < 8; ++r)
            ob[r] = (__bf16)(acc[d][r] * rinv);
        *(uint4*)&obuf[(size_t)t * DMODEL + h * DH + d * 16 + (hi ? 8 : 0)] =
            *(const uint4*)ob;
    }
}

// ---------------------------------------------------------------------------
// Launch: transposes -> QKV GEMM -> RoPE -> V^T -> flash attention -> out GEMM
// ---------------------------------------------------------------------------
extern "C" void kernel_launch(void* const* d_in, const int* in_sizes, int n_in,
                              void* d_out, int out_size, void* d_ws, size_t ws_size,
                              hipStream_t stream) {
    const __bf16* x     = (const __bf16*)d_in[0];   // (T, D) bf16
    const __bf16* Wqkv  = (const __bf16*)d_in[1];   // (D, 3D) bf16
    const __bf16* Wout  = (const __bf16*)d_in[2];   // (D, D) bf16
    const float*  cosb  = (const float*)d_in[3];    // (T, DH) f32
    const float*  sinb  = (const float*)d_in[4];    // (T, DH) f32
    const unsigned char* mask = (const unsigned char*)d_in[5];  // (T,) bool
    __bf16* out = (__bf16*)d_out;                   // (T, D) bf16

    char* ws = (char*)d_ws;
    size_t off = 0;
    __bf16* qkv    = (__bf16*)(ws + off); off += (size_t)TSEQ * D3 * 2;
    __bf16* WqkvT  = (__bf16*)(ws + off); off += (size_t)D3 * DMODEL * 2;
    __bf16* WoutT  = (__bf16*)(ws + off); off += (size_t)DMODEL * DMODEL * 2;
    __bf16* qrope  = (__bf16*)(ws + off); off += (size_t)TSEQ * DMODEL * 2;
    __bf16* krope  = (__bf16*)(ws + off); off += (size_t)TSEQ * DMODEL * 2;
    __bf16* vtrans = (__bf16*)(ws + off); off += (size_t)NH * DH * TSEQ * 2;
    __bf16* obuf   = (__bf16*)(ws + off); off += (size_t)TSEQ * DMODEL * 2;

    dim3 tb(32, 8);
    transpose_bf16<<<dim3(D3 / 32, DMODEL / 32, 1), tb, 0, stream>>>(
        Wqkv, WqkvT, DMODEL, D3, D3, DMODEL, 0, 0);
    transpose_bf16<<<dim3(DMODEL / 32, DMODEL / 32, 1), tb, 0, stream>>>(
        Wout, WoutT, DMODEL, DMODEL, DMODEL, DMODEL, 0, 0);

    gemm_bf16_wmma<<<dim3(D3 / 256, TSEQ / 128), 256, 0, stream>>>(
        x, WqkvT, qkv, TSEQ, D3, DMODEL);

    rope_kernel<<<(TSEQ * DMODEL) / 256, 256, 0, stream>>>(
        qkv, cosb, sinb, qrope, krope);

    transpose_bf16<<<dim3(DH / 32, TSEQ / 32, NH), tb, 0, stream>>>(
        qkv + 2 * DMODEL, vtrans, TSEQ, DH, D3, TSEQ,
        (long long)DH, (long long)DH * TSEQ);

    flash_attn<<<dim3(TSEQ / 64, NH), 128, 0, stream>>>(
        qrope, krope, vtrans, mask, obuf);

    gemm_bf16_wmma<<<dim3(DMODEL / 256, TSEQ / 128), 256, 0, stream>>>(
        obuf, WoutT, out, TSEQ, DMODEL, DMODEL);
}
